// GCNGRU_33509334843538
// MI455X (gfx1250) — matrix operand order repaired and checked by reading, not compiled, not run on hardware
//
#include <hip/hip_runtime.h>

// ---------------------------------------------------------------------------
// CDNA5 (gfx1250) wave32 WMMA implementation of the GCN seq2seq network.
// Matrix cores: v_wmma_f32_16x16x32_bf16 (fp32 data converted to bf16 once in
// LDS, fp32 accumulate). Main GEMM: 128-thread block (4 waves) computes a
// 16x64 C tile; A/W tiles staged in LDS via coalesced float4 global loads,
// fragments pulled with ds_load_b128. BN stats / softmax / K=3 convs: VALU.
// ---------------------------------------------------------------------------

typedef __attribute__((ext_vector_type(16))) __bf16 v16bf;
typedef __attribute__((ext_vector_type(8)))  __bf16 v8bf;
typedef __attribute__((ext_vector_type(8)))  float  v8f;

constexpr int   cB    = 128;
constexpr int   cN    = 48;
constexpr int   cF    = 256;
constexpr int   cTIN  = 10;
constexpr int   cTOUT = 10;
constexpr int   cNF   = cN * cF;    // 12288
constexpr int   cBN   = cB * cN;    // 6144
constexpr float cEPS   = 1e-5f;
constexpr float cSLOPE = 0.01f;

__device__ inline v8f wmma_bf16(v16bf a, v16bf b, v8f c) {
  // D = A(16x32 bf16) x B(32x16 bf16) + C(16x16 f32)
  return __builtin_amdgcn_wmma_f32_16x16x32_bf16(false, a, false, b,
                                                 (short)0, c, false, false);
}

__device__ inline v16bf cat8(v8bf lo, v8bf hi) {
  return __builtin_shufflevector(lo, hi, 0, 1, 2, 3, 4, 5, 6, 7,
                                         8, 9, 10, 11, 12, 13, 14, 15);
}

// ---------------------------------------------------------------------------
// Main GEMM: C[M,Nc] = concat_K(A0,A1)[M,Ktot] @ W[Ktot,Nc] (+bias)(+tanh)
// Block = 128 threads (4 waves); block tile 16(M) x 64(N); K stepped by 32.
// LDS: As[16][32] bf16 (row-major), Bs = W-tile transposed [64][40] bf16
//      (K contiguous per output column; row stride 40 keeps 16B alignment
//       for ds_load_b128 and breaks power-of-two bank conflicts).
// Fragment layouts (CDNA5 wave32):
//   A: half h=lane>>4, row r=lane&15; elem e<8 -> K=h*8+e, e>=8 -> K=16+h*8+e-8
//   B: col = lane&15, K same as A
//   C/D: col = lane&15, row = 8*h + elem
// ---------------------------------------------------------------------------
__global__ void __launch_bounds__(128)
k_gemm_bf16(const float* __restrict__ A0, int lda0,
            const float* __restrict__ A1, int lda1, int K0, int Ktot,
            const float* __restrict__ W, int ldw,
            const float* __restrict__ bias,
            float* __restrict__ C, int ldc, int mode)
{
  __shared__ __bf16 As[16 * 32];
  __shared__ __bf16 Bs[64 * 40];
  const int t    = threadIdx.x;
  const int lane = t & 31;
  const int wid  = t >> 5;          // wave id 0..3 -> N sub-tile
  const int half = lane >> 4;
  const int r    = lane & 15;
  const int m0   = blockIdx.x * 16;
  const int n0b  = blockIdx.y * 64;
  const int nw   = wid * 16;
  // A tile staging map: thread -> float4 at (row = t>>3, k = (t&7)*4)
  const int ar = t >> 3;
  const int ak = (t & 7) * 4;
  v8f acc = {0.f, 0.f, 0.f, 0.f, 0.f, 0.f, 0.f, 0.f};
  for (int k0 = 0; k0 < Ktot; k0 += 32) {
    // ---- stage A tile (16x32 f32 -> bf16), coalesced b128 loads ----
    const float* Arow = (k0 < K0) ? (A0 + (long)(m0 + ar) * lda0 + k0)
                                  : (A1 + (long)(m0 + ar) * lda1 + (k0 - K0));
    const float4 av = *(const float4*)(Arow + ak);
    As[ar * 32 + ak + 0] = (__bf16)av.x;
    As[ar * 32 + ak + 1] = (__bf16)av.y;
    As[ar * 32 + ak + 2] = (__bf16)av.z;
    As[ar * 32 + ak + 3] = (__bf16)av.w;
    // ---- stage W tile (32x64 f32 -> bf16, transposed), coalesced ----
#pragma unroll
    for (int i = 0; i < 4; ++i) {
      const int linear = (t + i * 128) * 4;
      const int kk = linear >> 6;       // 0..31
      const int nn = linear & 63;       // 0..60 step 4
      const float4 wv = *(const float4*)(W + (long)(k0 + kk) * ldw + n0b + nn);
      Bs[(nn + 0) * 40 + kk] = (__bf16)wv.x;
      Bs[(nn + 1) * 40 + kk] = (__bf16)wv.y;
      Bs[(nn + 2) * 40 + kk] = (__bf16)wv.z;
      Bs[(nn + 3) * 40 + kk] = (__bf16)wv.w;
    }
    __syncthreads();
    // ---- fragments: 2x ds_load_b128 each ----
    const __bf16* ap = As + r * 32 + half * 8;
    const v16bf a = cat8(*(const v8bf*)ap, *(const v8bf*)(ap + 16));
    const __bf16* bp = Bs + (nw + r) * 40 + half * 8;
    const v16bf b = cat8(*(const v8bf*)bp, *(const v8bf*)(bp + 16));
    acc = wmma_bf16(a, b, acc);
    __syncthreads();
  }
  const int n0 = n0b + nw;
  const float bv = bias ? bias[n0 + r] : 0.f;
#pragma unroll
  for (int e = 0; e < 8; ++e) {
    float v = acc[e] + bv;
    if (mode == 1) v = tanhf(v);
    C[(long)(m0 + half * 8 + e) * ldc + n0 + r] = v;
  }
}

// ---------------------------------------------------------------------------
// Batched node-mixing GEMM: Out[b] = Att_b[48,48] @ In[b][48,F] (+bias)
// attStride = 0 -> shared att (gconv); attStride = 48*48 -> per-batch (mix).
// grid = (B, 48/16, F/16), block = 32. K=48 padded to 64 with zeros.
// ---------------------------------------------------------------------------
__global__ void __launch_bounds__(32)
k_att_bf16(const float* __restrict__ AttBase, long attStride,
           const float* __restrict__ In,
           const float* __restrict__ bias,
           float* __restrict__ Out)
{
  const int lane = threadIdx.x;
  const int half = lane >> 4;
  const int r    = lane & 15;
  const int m0   = blockIdx.y * 16;
  const int n0   = blockIdx.z * 16;
  const float* att = AttBase + (long)blockIdx.x * attStride;
  const float* in  = In  + (long)blockIdx.x * cN * cF;
  float*       out = Out + (long)blockIdx.x * cN * cF;
  v8f acc = {0.f, 0.f, 0.f, 0.f, 0.f, 0.f, 0.f, 0.f};
  for (int k0 = 0; k0 < cN; k0 += 32) {
    v16bf a, b;
#pragma unroll
    for (int e = 0; e < 16; ++e) {
      const int ka = k0 + half * 16 + e;
      a[e] = (__bf16)((ka < cN) ? att[(m0 + r) * cN + ka] : 0.f);
      const int kb = k0 + half * 16 + e;
      b[e] = (__bf16)((kb < cN) ? in[(long)kb * cF + n0 + r] : 0.f);
    }
    acc = wmma_bf16(a, b, acc);
  }
  const float bv = bias ? bias[n0 + r] : 0.f;
#pragma unroll
  for (int e = 0; e < 8; ++e)
    out[(long)(m0 + half * 8 + e) * cF + n0 + r] = acc[e] + bv;
}

// ---------------------------------------------------------------------------
// Attention scores: S[b] = Q[b][48,256] @ Ctx[b][48,256]^T  -> [48,48]
// B fragment reads rows of Ctx (contiguous), since B[k][m] = Ctx[m][k].
// grid = (B, 3, 3), block = 32.
// ---------------------------------------------------------------------------
__global__ void __launch_bounds__(32)
k_scores_bf16(const float* __restrict__ Q, const float* __restrict__ Ctx,
              float* __restrict__ S)
{
  const int lane = threadIdx.x;
  const int half = lane >> 4;
  const int r    = lane & 15;
  const int m0   = blockIdx.y * 16;
  const int n0   = blockIdx.z * 16;
  const float* q = Q   + (long)blockIdx.x * cN * cF;
  const float* c = Ctx + (long)blockIdx.x * cN * cF;
  float*       s = S   + (long)blockIdx.x * cN * cN;
  v8f acc = {0.f, 0.f, 0.f, 0.f, 0.f, 0.f, 0.f, 0.f};
  for (int k0 = 0; k0 < cF; k0 += 32) {
    v16bf a, b;
#pragma unroll
    for (int e = 0; e < 16; ++e) {
      const int ka = k0 + half * 16 + e;
      a[e] = (__bf16)q[(long)(m0 + r) * cF + ka];
      const int kb = k0 + half * 16 + e;
      b[e] = (__bf16)c[(long)(n0 + r) * cF + kb];
    }
    acc = wmma_bf16(a, b, acc);
  }
#pragma unroll
  for (int e = 0; e < 8; ++e)
    s[(long)(m0 + half * 8 + e) * cN + n0 + r] = acc[e];
}

// -------------------------- VALU helper kernels ----------------------------

// support = x_frame[BN,3] @ gc1_w[3,F]
__global__ void k_support3(const float* __restrict__ X, int strideRow, int frameOff,
                           const float* __restrict__ W, float* __restrict__ Out)
{
  int idx = blockIdx.x * blockDim.x + threadIdx.x;
  if (idx >= cBN * cF) return;
  int row = idx >> 8;          // / cF
  int f   = idx & (cF - 1);
  const float* xr = X + (long)row * strideRow + frameOff;
  Out[idx] = xr[0] * W[f] + xr[1] * W[cF + f] + xr[2] * W[2 * cF + f];
}

// BatchNorm training-mode stats over the batch axis (biased variance).
__global__ void k_bn_stats(const float* __restrict__ X,
                           float* __restrict__ Mu, float* __restrict__ Var)
{
  int j = blockIdx.x * blockDim.x + threadIdx.x;
  if (j >= cNF) return;
  float s = 0.f, s2 = 0.f;
  for (int b = 0; b < cB; ++b) {
    float v = X[(long)b * cNF + j];
    s += v; s2 += v * v;
  }
  float mu = s * (1.f / cB);
  Mu[j]  = mu;
  Var[j] = fmaxf(s2 * (1.f / cB) - mu * mu, 0.f);
}

__global__ void k_bn_lrelu(const float* __restrict__ X, const float* __restrict__ Mu,
                           const float* __restrict__ Var, const float* __restrict__ G,
                           const float* __restrict__ Bt, float* __restrict__ Y)
{
  int idx = blockIdx.x * blockDim.x + threadIdx.x;
  if (idx >= cB * cNF) return;
  int j = idx % cNF;
  float y = (X[idx] - Mu[j]) * rsqrtf(Var[j] + cEPS) * G[j] + Bt[j];
  Y[idx] = (y >= 0.f) ? y : cSLOPE * y;
}

__global__ void k_add(const float* __restrict__ A, const float* __restrict__ B,
                      float* __restrict__ C, int n)
{
  int idx = blockIdx.x * blockDim.x + threadIdx.x;
  if (idx < n) C[idx] = A[idx] + B[idx];
}

__global__ void k_softmax48(float* __restrict__ S)
{
  int row = blockIdx.x * blockDim.x + threadIdx.x;
  if (row >= cB * cN) return;
  float* s = S + (long)row * cN;
  float mx = -3.4e38f;
  for (int k = 0; k < cN; ++k) mx = fmaxf(mx, s[k]);
  float sum = 0.f;
  for (int k = 0; k < cN; ++k) { float e = __expf(s[k] - mx); s[k] = e; sum += e; }
  float inv = 1.f / sum;
  for (int k = 0; k < cN; ++k) s[k] *= inv;
}

// T3[BN,3] = G1[BN,256] @ dec_w[256,3]
__global__ void k_decw(const float* __restrict__ G1, const float* __restrict__ W,
                       float* __restrict__ T3)
{
  int idx = blockIdx.x * blockDim.x + threadIdx.x;
  if (idx >= cBN * 3) return;
  int row = idx / 3, c3 = idx - row * 3;
  const float* g = G1 + (long)row * cF;
  float s = 0.f;
  for (int f = 0; f < cF; ++f) s += g[f] * W[f * 3 + c3];
  T3[idx] = s;
}

// OutF[b,n,c] = dec_att @ T3 + dec_b ; also scatter into d_out[b,n,t,c]
__global__ void k_decatt(const float* __restrict__ Att, const float* __restrict__ T3,
                         const float* __restrict__ Bias, float* __restrict__ OutF,
                         float* __restrict__ Y, int t)
{
  int idx = blockIdx.x * blockDim.x + threadIdx.x;
  if (idx >= cBN * 3) return;
  int c3 = idx % 3;
  int n  = (idx / 3) % cN;
  int b  = idx / (3 * cN);
  float s = Bias[c3];
  const float* tb = T3 + (long)b * cN * 3;
  for (int m = 0; m < cN; ++m) s += Att[n * cN + m] * tb[m * 3 + c3];
  OutF[idx] = s;
  Y[(((long)b * cN + n) * cTOUT + t) * 3 + c3] = s;
}

// ---------------------------------------------------------------------------
extern "C" void kernel_launch(void* const* d_in, const int* in_sizes, int n_in,
                              void* d_out, int out_size, void* d_ws, size_t ws_size,
                              hipStream_t stream)
{
  (void)in_sizes; (void)n_in; (void)out_size; (void)ws_size;
  const float* x        = (const float*)d_in[0];
  const float* gc1_w    = (const float*)d_in[1];
  const float* gc1_att  = (const float*)d_in[2];
  const float* gc1_b    = (const float*)d_in[3];
  const float* bn1_g    = (const float*)d_in[4];
  const float* bn1_b    = (const float*)d_in[5];
  const float* gcb_w1   = (const float*)d_in[6];
  const float* gcb_att1 = (const float*)d_in[7];
  const float* gcb_b1   = (const float*)d_in[8];
  const float* gcb_bn1g = (const float*)d_in[9];
  const float* gcb_bn1b = (const float*)d_in[10];
  const float* gcb_w2   = (const float*)d_in[11];
  const float* gcb_att2 = (const float*)d_in[12];
  const float* gcb_b2   = (const float*)d_in[13];
  const float* gcb_bn2g = (const float*)d_in[14];
  const float* gcb_bn2b = (const float*)d_in[15];
  const float* dec_w    = (const float*)d_in[16];
  const float* dec_att  = (const float*)d_in[17];
  const float* dec_b    = (const float*)d_in[18];
  const float* attn_w   = (const float*)d_in[19];
  const float* attn_b   = (const float*)d_in[20];

  float* ws = (float*)d_ws;
  const long S1 = (long)cBN * cF;        // 1.57M floats per activation buffer
  float* tmpA = ws;
  float* tmpB = tmpA + S1;
  float* F1   = tmpB + S1;
  float* Zb   = F1   + S1;
  float* G1a  = Zb   + S1;
  float* G1b  = G1a  + S1;
  float* MIX  = G1b  + S1;
  float* Ssc  = MIX  + S1;               // [B,48,48]
  float* Mu   = Ssc  + (long)cB * cN * cN;
  float* Varv = Mu   + cNF;
  float* T3   = Varv + cNF;              // [BN,3]
  float* OutF = T3   + (long)cBN * 3;    // [BN,3]

  const dim3 gGemm(cBN / 16, cF / 64);   // 16x64 block tiles, 4 waves/block
  const dim3 gAtt(cB, cN / 16, cF / 16);
  const dim3 gSc(cB, cN / 16, cN / 16);
  const int  eBNF = cBN * cF;

  auto bnpair = [&](const float* in, const float* g, const float* bt, float* outp) {
    k_bn_stats<<<(cNF + 127) / 128, 128, 0, stream>>>(in, Mu, Varv);
    k_bn_lrelu<<<(eBNF + 255) / 256, 256, 0, stream>>>(in, Mu, Varv, g, bt, outp);
  };
  auto embed = [&](const float* Xb, int strideRow, int frameOff) {
    k_support3<<<(eBNF + 255) / 256, 256, 0, stream>>>(Xb, strideRow, frameOff, gc1_w, tmpA);
    k_att_bf16<<<gAtt, 32, 0, stream>>>(gc1_att, 0L, tmpA, gc1_b, tmpB);
    bnpair(tmpB, bn1_g, bn1_b, F1);
  };
  auto gcb = [&](int i, const float* a, const float* b_in, float* gout) {
    k_gemm_bf16<<<gGemm, 128, 0, stream>>>(a, cF, a, cF, cF, cF,
        gcb_w1 + (long)i * cF * cF, cF, (const float*)nullptr, tmpA, cF, 0);
    k_att_bf16<<<gAtt, 32, 0, stream>>>(gcb_att1 + (long)i * cN * cN, 0L, tmpA,
                                        gcb_b1 + (long)i * cF, tmpB);
    bnpair(tmpB, gcb_bn1g + (long)i * cNF, gcb_bn1b + (long)i * cNF, Zb);
    k_add<<<(eBNF + 255) / 256, 256, 0, stream>>>(Zb, b_in, tmpA, eBNF);
    k_gemm_bf16<<<gGemm, 128, 0, stream>>>(tmpA, cF, tmpA, cF, cF, cF,
        gcb_w2 + (long)i * cF * cF, cF, (const float*)nullptr, tmpB, cF, 0);
    k_att_bf16<<<gAtt, 32, 0, stream>>>(gcb_att2 + (long)i * cN * cN, 0L, tmpB,
                                        gcb_b2 + (long)i * cF, Zb);
    bnpair(Zb, gcb_bn2g + (long)i * cNF, gcb_bn2b + (long)i * cNF, gout);
  };

  // ---------------- encoder: recurrent accumulation over input frames ------
  for (int i = 0; i < cTIN; ++i) {
    embed(x, cTIN * 3, i * 3);                         // F1 = embed(x[:,:,i,:])
    gcb(i, (i == 0) ? F1 : G1a, F1, G1a);              // g1 = gcb(i, ., f1)
  }

  // ---------------- decoder: autoregressive output frames ------------------
  for (int i = 0; i < cTOUT; ++i) {
    k_decw<<<(cBN * 3 + 255) / 256, 256, 0, stream>>>(G1a, dec_w + (long)i * cF * 3, T3);
    k_decatt<<<(cBN * 3 + 255) / 256, 256, 0, stream>>>(dec_att + (long)i * cN * cN, T3,
        dec_b + (long)i * 3, OutF, (float*)d_out, i);
    embed(OutF, 3, 0);                                 // F1 = embed(outF)
    // Luong attention over nodes
    k_scores_bf16<<<gSc, 32, 0, stream>>>(G1a, F1, Ssc);
    k_softmax48<<<(cBN + 255) / 256, 256, 0, stream>>>(Ssc);
    k_att_bf16<<<gAtt, 32, 0, stream>>>(Ssc, (long)cN * cN, F1,
                                        (const float*)nullptr, MIX);
    // tanh(concat(mix, g1) @ attn_w + attn_b) without materializing concat
    k_gemm_bf16<<<gGemm, 128, 0, stream>>>(MIX, cF, G1a, cF, cF, 2 * cF,
        attn_w, cF, attn_b, G1b, cF, 1);
    gcb(cTIN + i, G1b, F1, G1a);
  }
}
